// NTXentLoss_29841432773048
// MI455X (gfx1250) — compile-verified, matrix operand order checked by reading, not run on hardware
//
#include <hip/hip_runtime.h>
#include <hip/hip_bf16.h>

#define N_    8192
#define D_    256
#define INV_T 1.25f      // 1 / 0.8
#define EPS_  1e-7f

typedef __attribute__((ext_vector_type(16))) __bf16 v16bf;
typedef __attribute__((ext_vector_type(8)))  __bf16 v8bf;
typedef __attribute__((ext_vector_type(8)))  float  v8f;

union Frag { v16bf v; v8bf h[2]; };

static __device__ __forceinline__ unsigned short f32_to_bf16(float x) {
  __hip_bfloat16 h = __float2bfloat16(x);           // RNE
  return __builtin_bit_cast(unsigned short, h);
}
static __device__ __forceinline__ float bf16_to_f32(unsigned short u) {
  __hip_bfloat16 h = __builtin_bit_cast(__hip_bfloat16, u);
  return __bfloat162float(h);
}

// ---------------------------------------------------------------------------
// Kernel 1: L2-normalize each row, emit bf16 hi/lo split (x ~= hi + lo).
// One block per row, one thread per element (D_ == 256).
// ---------------------------------------------------------------------------
__global__ __launch_bounds__(256) void normalize_split_kernel(
    const float* __restrict__ feat,
    unsigned short* __restrict__ fhi,
    unsigned short* __restrict__ flo)
{
  const int row = blockIdx.x;
  const int t   = threadIdx.x;
  const float x = feat[row * D_ + t];

  float s = x * x;
  #pragma unroll
  for (int m = 16; m >= 1; m >>= 1) s += __shfl_xor(s, m, 32);

  __shared__ float sm[8];
  __shared__ float s_inv;
  if ((t & 31) == 0) sm[t >> 5] = s;
  __syncthreads();
  if (t == 0) {
    float tot = 0.f;
    #pragma unroll
    for (int i = 0; i < 8; ++i) tot += sm[i];
    s_inv = 1.0f / fmaxf(__builtin_sqrtf(tot), 1e-12f);
  }
  __syncthreads();

  const float xn = x * s_inv;
  const unsigned short h = f32_to_bf16(xn);
  const unsigned short l = f32_to_bf16(xn - bf16_to_f32(h));
  fhi[row * D_ + t] = h;
  flo[row * D_ + t] = l;
}

// ---------------------------------------------------------------------------
// Kernel 2: fused Gram-matrix (bf16x3 split WMMA) + exp + masked row sums.
// Grid: N/128 workgroups; 8 waves/WG; wave owns a 16-row strip, sweeps all
// 512 column tiles. Per K-chunk: 3x v_wmma_f32_16x16x32_bf16.
// ---------------------------------------------------------------------------
__global__ __launch_bounds__(256) void gram_loss_kernel(
    const unsigned short* __restrict__ fhi,
    const unsigned short* __restrict__ flo,
    const int* __restrict__ labels,
    const int* __restrict__ spk,
    float* __restrict__ partials)   // [gridDim.x][2] = {loss_sum, valid_count}
{
  const int lane = threadIdx.x & 31;
  const int wave = threadIdx.x >> 5;
  const int l16  = lane & 15;
  const int sel  = lane >> 4;                 // 0: K 0..7 & 16..23 ; 1: 8..15 & 24..31
  const int rowbase = blockIdx.x * 128 + wave * 16;

  // A fragments for this wave's 16-row strip (held in VGPRs for whole sweep)
  v16bf a_hi[8], a_lo[8];
  {
    const int arow = rowbase + l16;
    const unsigned short* ph = fhi + (size_t)arow * D_;
    const unsigned short* pl = flo + (size_t)arow * D_;
    #pragma unroll
    for (int kc = 0; kc < 8; ++kc) {
      const int k0 = kc * 32 + sel * 8;
      Frag fh, fl;
      fh.h[0] = *(const v8bf*)(ph + k0);
      fh.h[1] = *(const v8bf*)(ph + k0 + 16);
      fl.h[0] = *(const v8bf*)(pl + k0);
      fl.h[1] = *(const v8bf*)(pl + k0 + 16);
      a_hi[kc] = fh.v;
      a_lo[kc] = fl.v;
    }
  }

  // Labels/speakers of my 8 rows (C layout: M = k + 8*sel)
  int labr[8], spkr[8];
  #pragma unroll
  for (int k = 0; k < 8; ++k) {
    const int r = rowbase + k + 8 * sel;
    labr[k] = labels[r];
    spkr[k] = spk[r];
  }

  float numk[8], negk[8];
  #pragma unroll
  for (int k = 0; k < 8; ++k) { numk[k] = 0.f; negk[k] = 0.f; }

  for (int jt = 0; jt < N_ / 16; ++jt) {
    const int col  = jt * 16 + l16;           // B fragment: lane <-> N
    const int labc = labels[col];
    const int spkc = spk[col];
    const unsigned short* ph = fhi + (size_t)col * D_;
    const unsigned short* pl = flo + (size_t)col * D_;

    v8f c0 = {};                              // hi*hi chain
    v8f c1 = {};                              // cross-term chain (indep. for ILP)
    #pragma unroll
    for (int kc = 0; kc < 8; ++kc) {
      const int k0 = kc * 32 + sel * 8;
      Frag bh, bl;
      bh.h[0] = *(const v8bf*)(ph + k0);
      bh.h[1] = *(const v8bf*)(ph + k0 + 16);
      bl.h[0] = *(const v8bf*)(pl + k0);
      bl.h[1] = *(const v8bf*)(pl + k0 + 16);
      c0 = __builtin_amdgcn_wmma_f32_16x16x32_bf16(false, a_hi[kc], false, bh.v,
                                                   (short)0, c0, false, false);
      c1 = __builtin_amdgcn_wmma_f32_16x16x32_bf16(false, a_hi[kc], false, bl.v,
                                                   (short)0, c1, false, false);
      c1 = __builtin_amdgcn_wmma_f32_16x16x32_bf16(false, a_lo[kc], false, bh.v,
                                                   (short)0, c1, false, false);
    }

    // Fused epilogue: exp + masks + per-row accumulation (stays in registers)
    #pragma unroll
    for (int k = 0; k < 8; ++k) {
      const float e  = __expf((c0[k] + c1[k]) * INV_T);
      const int   i  = rowbase + k + 8 * sel;
      const bool  sl = (labc == labr[k]) && (i != col);
      const bool  ss = (spkc == spkr[k]);
      numk[k] += (sl && ss)  ? e : 0.0f;      // same emotion, same speaker
      negk[k] += (sl && !ss) ? e : 0.0f;      // same emotion, diff speaker
    }
  }

  // Row sums: reduce over the 16 lanes of each half (masks 1,2,4,8 stay in-half)
  #pragma unroll
  for (int k = 0; k < 8; ++k) {
    #pragma unroll
    for (int m = 8; m >= 1; m >>= 1) {
      numk[k] += __shfl_xor(numk[k], m, 32);
      negk[k] += __shfl_xor(negk[k], m, 32);
    }
  }

  __shared__ float s_ls[16], s_ct[16];
  if (l16 == 0) {                             // lanes 0 and 16: rows k+8*sel
    float ls = 0.f, ct = 0.f;
    #pragma unroll
    for (int k = 0; k < 8; ++k) {
      const float num = numk[k];
      if (num > 0.f) {                        // valid <=> has a positive (exp > 0)
        const float den = negk[k] + num + EPS_;
        ls += __logf(den) - __logf(num);      // -log(num/den)
        ct += 1.0f;
      }
    }
    s_ls[wave * 2 + sel] = ls;
    s_ct[wave * 2 + sel] = ct;
  }
  __syncthreads();
  if (threadIdx.x == 0) {                     // fixed-order -> deterministic
    float ls = 0.f, ct = 0.f;
    #pragma unroll
    for (int i = 0; i < 16; ++i) { ls += s_ls[i]; ct += s_ct[i]; }
    partials[2 * blockIdx.x]     = ls;
    partials[2 * blockIdx.x + 1] = ct;
  }
}

// ---------------------------------------------------------------------------
// Kernel 3: deterministic final reduction over 64 workgroup partials.
// ---------------------------------------------------------------------------
__global__ void finalize_kernel(const float* __restrict__ partials,
                                float* __restrict__ out)
{
  if (threadIdx.x == 0 && blockIdx.x == 0) {
    float tot = 0.f, ct = 0.f;
    for (int i = 0; i < N_ / 128; ++i) {
      tot += partials[2 * i];
      ct  += partials[2 * i + 1];
    }
    out[0] = (ct > 0.f) ? (tot / ct) : 0.f;
  }
}

extern "C" void kernel_launch(void* const* d_in, const int* in_sizes, int n_in,
                              void* d_out, int out_size, void* d_ws, size_t ws_size,
                              hipStream_t stream) {
  const float* feat   = (const float*)d_in[0];
  const int*   labels = (const int*)d_in[1];
  const int*   spk    = (const int*)d_in[2];

  // Workspace layout: bf16 hi (4 MB) | bf16 lo (4 MB) | 64x2 f32 partials
  unsigned short* fhi = (unsigned short*)d_ws;
  unsigned short* flo = fhi + (size_t)N_ * D_;
  float* partials     = (float*)(flo + (size_t)N_ * D_);

  normalize_split_kernel<<<N_, D_, 0, stream>>>(feat, fhi, flo);
  gram_loss_kernel<<<N_ / 128, 256, 0, stream>>>(fhi, flo, labels, spk, partials);
  finalize_kernel<<<1, 1, 0, stream>>>(partials, (float*)d_out);
}
